// CTM_72413148611050
// MI455X (gfx1250) — matrix operand order repaired and checked by reading, not compiled, not run on hardware
//
#include <hip/hip_runtime.h>
#include <hip/hip_bf16.h>
#include <math.h>

// ---------------------------------------------------------------------------
// Problem constants (match the JAX reference)
// ---------------------------------------------------------------------------
#define Bn 8
#define Hh 56
#define Wd 56
#define Nn 3136          // H*W
#define Ee 128           // EMBED
#define Dd 256           // D
#define Pp 784           // HP*WP (28*28)
#define Mm 784           // num cluster centers
#define K9 1152          // E * 3 * 3 (im2col K)

typedef __attribute__((ext_vector_type(16))) _Float16 v16h;
typedef __attribute__((ext_vector_type(8)))  float    v8f;
typedef __attribute__((ext_vector_type(4)))  unsigned int su4;
typedef __attribute__((ext_vector_type(8)))  unsigned int su8;

// LDS tile geometry: K-step 64 halves (128 B) + 16 B pad -> 144 B row stride
#define LDS_ROW_H   72            // halves per LDS row (64 data + 8 pad)
#define LDS_ROW_B   144           // bytes per LDS row
#define A_ROWS      128
#define B_ROWS      64
#define A_BYTES     (A_ROWS * LDS_ROW_B)   // 18432
#define B_BYTES     (B_ROWS * LDS_ROW_B)   // 9216

// ---------------------------------------------------------------------------
// Small utility kernels
// ---------------------------------------------------------------------------
__global__ void zerof_kernel(float* p, int n) {
  int i = blockIdx.x * 256 + threadIdx.x;
  if (i < n) p[i] = 0.f;
}

__global__ void cvt_f32_f16_kernel(const float* __restrict__ s, _Float16* __restrict__ d, int n) {
  int i = blockIdx.x * 256 + threadIdx.x;
  if (i < n) d[i] = (_Float16)s[i];
}

__device__ __forceinline__ float blockReduceSum(float v, float* red, int tid) {
  __syncthreads();
  red[tid] = v;
  __syncthreads();
  for (int off = 128; off > 0; off >>= 1) {
    if (tid < off) red[tid] += red[tid + off];
    __syncthreads();
  }
  return red[0];
}

__device__ __forceinline__ float blockReduceMax(float v, float* red, int tid) {
  __syncthreads();
  red[tid] = v;
  __syncthreads();
  for (int off = 128; off > 0; off >>= 1) {
    if (tid < off) red[tid] = fmaxf(red[tid], red[tid + off]);
    __syncthreads();
  }
  return red[0];
}

// ---------------------------------------------------------------------------
// Tensor Data Mover staging (CDNA5 ISA).
// 2D tile: (rows_slice x 64 halves) from a row-major (rows x K) f16 matrix,
// TDM inserts 4 DWORDs of LDS padding after every 32 DWORDs (row = 144 B).
// Group-1 descriptor is loop-invariant; group-0 carries the moving address.
// The LDS destination pointer is passed to the asm so the buffer escapes and
// the "memory" clobber legally covers the DMA writes (keeps ds_loads alive).
// ---------------------------------------------------------------------------
__device__ __forceinline__ su8 tdm_make_g1(int K, int rem_rows, int rows_slice) {
  unsigned td1 = (unsigned)(rem_rows < 0 ? 0 : rem_rows);
  su8 g1;
  g1[0] = (1u << 16)      // data_size = 2 bytes
        | (1u << 20)      // pad_enable
        | (4u << 22)      // pad_interval: 32 DWORDs (128 B of data)
        | (3u << 25);     // pad_amount: 4 DWORDs (16 B)
  g1[1] = ((unsigned)K & 0xffffu) << 16;                             // tensor_dim0 lo
  g1[2] = (((unsigned)K >> 16) & 0xffffu) | ((td1 & 0xffffu) << 16); // dim0 hi | dim1 lo
  g1[3] = ((td1 >> 16) & 0xffffu) | (64u << 16);                     // dim1 hi | tile_dim0=64
  g1[4] = (unsigned)rows_slice & 0xffffu;                            // tile_dim1
  g1[5] = (unsigned)K;                                               // tensor_dim0_stride lo
  g1[6] = 0u;
  g1[7] = 0u;
  return g1;
}

__device__ __forceinline__ void tdm_issue(void* lds_ptr, unsigned long long ga, su8 g1) {
  su4 g0;
  g0[0] = 1u;                                                  // count=1 (valid user D#)
  g0[1] = (unsigned)(unsigned long long)lds_ptr;               // lds_addr = addr[31:0]
  g0[2] = (unsigned)(ga & 0xffffffffull);                      // global_addr[31:0]
  g0[3] = (unsigned)((ga >> 32) & 0x1ffffffull) | (2u << 30);  // global_addr[56:32], type=2
  asm volatile("tensor_load_to_lds %0, %1"
               :: "s"(g0), "s"(g1), "s"(lds_ptr)
               : "memory");
}

// ---------------------------------------------------------------------------
// WMMA fragment math (CDNA5 layouts, wave32; CDNA5 ISA §7.12.2)
//   A 16x32 f16 : lane l (m=l&15) holds K {base..base+7, 16+base..16+base+7}
//   B 32x16 f16 : lane l (n=l&15) holds K {kb..kb+7, 16+kb..16+kb+7}, kb=(l<16)?0:8
//   C 16x16 f32 : lane l (n=l&15), vgpr q -> row q + ((l<16)?0:8)
// ---------------------------------------------------------------------------
__device__ __forceinline__ void wmma_tile_2x2(const _Float16* Ash, const _Float16* Bsh,
                                              int kc, int lane, int wm, int wn,
                                              v8f acc[2][2]) {
  v16h af[2], bf[2];
  const int base = (lane < 16) ? 0 : 8;
  const int kb   = (lane < 16) ? 0 : 8;
#pragma unroll
  for (int mi = 0; mi < 2; ++mi) {
    const _Float16* ap = Ash + (wm * 32 + mi * 16 + (lane & 15)) * LDS_ROW_H + kc;
#pragma unroll
    for (int q = 0; q < 8; ++q) {
      af[mi][q]     = ap[base + q];
      af[mi][8 + q] = ap[16 + base + q];
    }
  }
#pragma unroll
  for (int ni = 0; ni < 2; ++ni) {
    const _Float16* bp = Bsh + (wn * 32 + ni * 16 + (lane & 15)) * LDS_ROW_H + kc;
#pragma unroll
    for (int q = 0; q < 8; ++q) {
      bf[ni][q]     = bp[kb + q];
      bf[ni][8 + q] = bp[16 + kb + q];
    }
  }
#pragma unroll
  for (int mi = 0; mi < 2; ++mi)
#pragma unroll
    for (int ni = 0; ni < 2; ++ni)
      acc[mi][ni] = __builtin_amdgcn_wmma_f32_16x16x32_f16(
          false, af[mi], false, bf[ni], (short)0, acc[mi][ni], false, false);
}

// ---------------------------------------------------------------------------
// Generic tiled GEMM: C[M x N] = A(MxK,f16) * B(NxK,f16)^T (+bias[N])
// Block 256 threads (8 waves), tile 128(M) x 64(N), K step 64 (2 WMMA chunks).
// Tiles staged by the Tensor Data Mover: each wave DMAs its own row slice.
// ---------------------------------------------------------------------------
__global__ __launch_bounds__(256) void wmma_gemm_kernel(
    const _Float16* __restrict__ A, const _Float16* __restrict__ Bm,
    const float* __restrict__ bias, float* __restrict__ C,
    int Mtot, int Ntot, int Ktot) {
  __shared__ __align__(16) unsigned char lds_raw[A_BYTES + B_BYTES];
  _Float16* Ash = (_Float16*)lds_raw;
  _Float16* Bsh = (_Float16*)(lds_raw + A_BYTES);
  const int tid = threadIdx.x, lane = tid & 31;
  const int wv = __builtin_amdgcn_readfirstlane(tid >> 5);   // wave id 0..7 (scalar)
  const int wm = wv & 3, wn = wv >> 2;
  const int m0 = blockIdx.y * 128, n0 = blockIdx.x * 64;

  // per-wave DMA slices (A: 16 rows, B: 8 rows), loop-invariant descriptors
  int r0a = m0 + wv * 16, rema = Mtot - r0a; if (rema <= 0) r0a = 0;
  int r0b = n0 + wv * 8,  remb = Ntot - r0b; if (remb <= 0) r0b = 0;
  void* ldsA = (void*)(lds_raw + wv * 16 * LDS_ROW_B);
  void* ldsB = (void*)(lds_raw + A_BYTES + wv * 8 * LDS_ROW_B);
  su8 g1a = tdm_make_g1(Ktot, rema, 16);
  su8 g1b = tdm_make_g1(Ktot, remb, 8);
  unsigned long long gaA = (unsigned long long)A +
                           (unsigned long long)(unsigned)r0a * (unsigned)Ktot * 2ull;
  unsigned long long gaB = (unsigned long long)Bm +
                           (unsigned long long)(unsigned)r0b * (unsigned)Ktot * 2ull;

  v8f acc[2][2] = {};
  for (int k0 = 0; k0 < Ktot; k0 += 64) {
    tdm_issue(ldsA, gaA, g1a);
    tdm_issue(ldsB, gaB, g1b);
    __builtin_amdgcn_s_wait_tensorcnt(0);
    __syncthreads();
    wmma_tile_2x2(Ash, Bsh, 0, lane, wm, wn, acc);
    wmma_tile_2x2(Ash, Bsh, 32, lane, wm, wn, acc);
    __syncthreads();
    gaA += 128ull;   // 64 halves per K step
    gaB += 128ull;
  }
#pragma unroll
  for (int mi = 0; mi < 2; ++mi) {
#pragma unroll
    for (int ni = 0; ni < 2; ++ni) {
      int gn = n0 + wn * 32 + ni * 16 + (lane & 15);
      int mb = m0 + wm * 32 + mi * 16 + ((lane < 16) ? 0 : 8);
      if (gn < Ntot) {
        float bv = bias ? bias[gn] : 0.f;
#pragma unroll
        for (int q = 0; q < 8; ++q) {
          int gm = mb + q;
          if (gm < Mtot) C[(size_t)gm * Ntot + gn] = acc[mi][ni][q] + bv;
        }
      }
    }
  }
}

// ---------------------------------------------------------------------------
// Pairwise distance GEMM for one batch:
//   dist[i][j] = sqrt(max(x2[i]+x2[j]-2*<xh_i,xh_j>, 0)) / 16   (sqrt(D)=16)
//   also: atomic max of dist into *dmaxp (nonneg float -> int atomicMax trick)
// ---------------------------------------------------------------------------
__global__ __launch_bounds__(256) void wmma_dist_kernel(
    const _Float16* __restrict__ xh, const float* __restrict__ x2,
    float* __restrict__ dist, float* __restrict__ dmaxp) {
  __shared__ __align__(16) unsigned char lds_raw[A_BYTES + B_BYTES + 1024];
  _Float16* Ash = (_Float16*)lds_raw;
  _Float16* Bsh = (_Float16*)(lds_raw + A_BYTES);
  float*    red = (float*)(lds_raw + A_BYTES + B_BYTES);
  const int tid = threadIdx.x, lane = tid & 31;
  const int wv = __builtin_amdgcn_readfirstlane(tid >> 5);
  const int wm = wv & 3, wn = wv >> 2;
  const int m0 = blockIdx.y * 128, n0 = blockIdx.x * 64;

  int r0a = m0 + wv * 16, rema = Nn - r0a; if (rema <= 0) r0a = 0;
  int r0b = n0 + wv * 8,  remb = Nn - r0b; if (remb <= 0) r0b = 0;
  void* ldsA = (void*)(lds_raw + wv * 16 * LDS_ROW_B);
  void* ldsB = (void*)(lds_raw + A_BYTES + wv * 8 * LDS_ROW_B);
  su8 g1a = tdm_make_g1(Dd, rema, 16);
  su8 g1b = tdm_make_g1(Dd, remb, 8);
  unsigned long long gaA = (unsigned long long)xh +
                           (unsigned long long)(unsigned)r0a * (unsigned)Dd * 2ull;
  unsigned long long gaB = (unsigned long long)xh +
                           (unsigned long long)(unsigned)r0b * (unsigned)Dd * 2ull;

  v8f acc[2][2] = {};
  for (int k0 = 0; k0 < Dd; k0 += 64) {
    tdm_issue(ldsA, gaA, g1a);
    tdm_issue(ldsB, gaB, g1b);
    __builtin_amdgcn_s_wait_tensorcnt(0);
    __syncthreads();
    wmma_tile_2x2(Ash, Bsh, 0, lane, wm, wn, acc);
    wmma_tile_2x2(Ash, Bsh, 32, lane, wm, wn, acc);
    __syncthreads();
    gaA += 128ull;
    gaB += 128ull;
  }
  float tmax = 0.f;
#pragma unroll
  for (int mi = 0; mi < 2; ++mi) {
#pragma unroll
    for (int ni = 0; ni < 2; ++ni) {
      int gn = n0 + wn * 32 + ni * 16 + (lane & 15);
      int mb = m0 + wm * 32 + mi * 16 + ((lane < 16) ? 0 : 8);
      if (gn < Nn) {
        float xj = x2[gn];
#pragma unroll
        for (int q = 0; q < 8; ++q) {
          int gm = mb + q;
          if (gm < Nn) {
            float v  = x2[gm] + xj - 2.f * acc[mi][ni][q];
            float dd = sqrtf(fmaxf(v, 0.f)) * 0.0625f;
            dist[(size_t)gm * Nn + gn] = dd;
            tmax = fmaxf(tmax, dd);
          }
        }
      }
    }
  }
  float bm = blockReduceMax(tmax, red, tid);
  if (tid == 0) atomicMax((int*)dmaxp, __float_as_int(bm));
}

// ---------------------------------------------------------------------------
// im2col for the 3x3 stride-2 pad-1 conv: col[b*P+p][e*9+kh*3+kw] (f16)
// ---------------------------------------------------------------------------
__global__ __launch_bounds__(256) void im2col_kernel(const float* __restrict__ x,
                                                     _Float16* __restrict__ col) {
  int r = blockIdx.x;            // b*Pp + p
  int b = r / Pp, p = r % Pp;
  int hp = p / 28, wp = p % 28;
  for (int k = threadIdx.x; k < K9; k += 256) {
    int e = k / 9, rem = k % 9, kh = rem / 3, kw = rem % 3;
    int h = 2 * hp - 1 + kh, w2 = 2 * wp - 1 + kw;
    float v = 0.f;
    if (h >= 0 && h < Hh && w2 >= 0 && w2 < Wd)
      v = x[((size_t)b * Nn + h * Wd + w2) * Ee + e];
    col[(size_t)r * K9 + k] = (_Float16)v;
  }
}

// ---------------------------------------------------------------------------
// xc = xsk + gather(x3) ; xn = LayerNorm(xc) ; token_score ; token_weight
// ---------------------------------------------------------------------------
__global__ __launch_bounds__(256) void ln_xc_kernel(
    const float* __restrict__ xsk, const float* __restrict__ x3,
    const float* __restrict__ g, const float* __restrict__ bta,
    const float* __restrict__ sw, const float* __restrict__ sb,
    float* __restrict__ xn, _Float16* __restrict__ xnh,
    float* __restrict__ tscore, float* __restrict__ tw) {
  __shared__ float red[256];
  int gi = blockIdx.x;           // b*Nn + n
  int b = gi / Nn, n = gi % Nn;
  int h = n / Wd, w = n % Wd;
  int gp = (h >> 1) * 28 + (w >> 1);
  int tid = threadIdx.x;
  float v  = xsk[(size_t)gi * Dd + tid] + x3[((size_t)(b * Pp + gp)) * Dd + tid];
  float mu = blockReduceSum(v, red, tid) * (1.f / 256.f);
  float dv = v - mu;
  float var = blockReduceSum(dv * dv, red, tid) * (1.f / 256.f);
  float y = dv * rsqrtf(var + 1e-5f) * g[tid] + bta[tid];
  xn[(size_t)gi * Dd + tid]  = y;
  xnh[(size_t)gi * Dd + tid] = (_Float16)y;
  float sc = blockReduceSum(y * sw[tid], red, tid);
  if (tid == 0) {
    float s = sc + sb[0];
    tscore[gi] = s;
    tw[gi] = expf(s);
  }
}

// x2[r] = sum_d xn[r][d]^2
__global__ __launch_bounds__(256) void x2_kernel(const float* __restrict__ xn, float* __restrict__ x2) {
  __shared__ float red[256];
  int r = blockIdx.x, tid = threadIdx.x;
  float v = xn[(size_t)r * Dd + tid];
  float s = blockReduceSum(v * v, red, tid);
  if (tid == 0) x2[r] = s;
}

// ---------------------------------------------------------------------------
// density: exp(-mean of 5 smallest dist^2 per row) + tiny deterministic noise
// ---------------------------------------------------------------------------
__global__ __launch_bounds__(256) void density_kernel(const float* __restrict__ dist,
                                                      float* __restrict__ dens, int b) {
  __shared__ float sm[256 * 5];
  int i = blockIdx.x, tid = threadIdx.x;
  float t[5];
#pragma unroll
  for (int q = 0; q < 5; ++q) t[q] = 3.0e38f;
  const float* row = dist + (size_t)i * Nn;
  for (int j = tid; j < Nn; j += 256) {
    __builtin_prefetch(row + j + 1024, 0, 0);
    float v = row[j];
    if (v < t[4]) {
      t[4] = v;
#pragma unroll
      for (int q = 4; q > 0; --q) {
        if (t[q] < t[q - 1]) { float tmp = t[q]; t[q] = t[q - 1]; t[q - 1] = tmp; }
      }
    }
  }
#pragma unroll
  for (int q = 0; q < 5; ++q) sm[tid * 5 + q] = t[q];
  __syncthreads();
  for (int off = 128; off > 0; off >>= 1) {
    if (tid < off) {
      float* a = &sm[tid * 5];
      float* c = &sm[(tid + off) * 5];
      float o[5]; int ia = 0, ib = 0;
#pragma unroll
      for (int q = 0; q < 5; ++q) {
        if (a[ia] <= c[ib]) o[q] = a[ia++]; else o[q] = c[ib++];
      }
#pragma unroll
      for (int q = 0; q < 5; ++q) a[q] = o[q];
    }
    __syncthreads();
  }
  if (tid == 0) {
    float mean = 0.f;
#pragma unroll
    for (int q = 0; q < 5; ++q) mean += sm[q] * sm[q];
    mean *= (1.f / 5.f);
    unsigned hsh = (unsigned)(b * Nn + i);
    hsh = hsh * 2654435761u ^ 0x9E3779B9u; hsh ^= hsh >> 16; hsh *= 0x85EBCA6Bu; hsh ^= hsh >> 13;
    float u = (float)(hsh & 0xFFFFFFu) * (1.0f / 16777216.0f);
    dens[(size_t)b * Nn + i] = expf(-mean) + u * 1e-6f;
  }
}

// dmin[i] = min_{j: dens[j]>dens[i]} dist[i][j] (init = dist_max) ; score = dmin*dens
__global__ __launch_bounds__(256) void dmin_kernel(const float* __restrict__ dist,
                                                   const float* __restrict__ dens,
                                                   const float* __restrict__ dmaxp,
                                                   float* __restrict__ score, int b) {
  __shared__ float sdens[Nn];
  __shared__ float red[256];
  int i = blockIdx.x, tid = threadIdx.x;
  const float* db = dens + (size_t)b * Nn;
  for (int j = tid; j < Nn; j += 256) sdens[j] = db[j];
  __syncthreads();
  float di = sdens[i];
  float local = dmaxp[0];
  const float* row = dist + (size_t)i * Nn;
  for (int j = tid; j < Nn; j += 256) {
    __builtin_prefetch(row + j + 1024, 0, 0);
    if (sdens[j] > di) local = fminf(local, row[j]);
  }
  red[tid] = local;
  __syncthreads();
  for (int off = 128; off > 0; off >>= 1) {
    if (tid < off) red[tid] = fminf(red[tid], red[tid + off]);
    __syncthreads();
  }
  if (tid == 0) score[(size_t)b * Nn + i] = red[0] * di;
}

// iterative top-M selection (descending, ties -> lower index); one block per batch
__global__ __launch_bounds__(256) void topm_kernel(const float* __restrict__ score,
                                                   int* __restrict__ idxdown, int b) {
  __shared__ float sv[Nn];
  __shared__ float rv[256];
  __shared__ int   ri[256];
  int tid = threadIdx.x;
  for (int j = tid; j < Nn; j += 256) sv[j] = score[(size_t)b * Nn + j];
  __syncthreads();
  for (int it = 0; it < Mm; ++it) {
    float bv = -3.0e38f; int bi = 0x7fffffff;
    for (int j = tid; j < Nn; j += 256) {
      float v = sv[j];
      if (v > bv || (v == bv && j < bi)) { bv = v; bi = j; }
    }
    rv[tid] = bv; ri[tid] = bi;
    __syncthreads();
    for (int off = 128; off > 0; off >>= 1) {
      if (tid < off) {
        if (rv[tid + off] > rv[tid] ||
            (rv[tid + off] == rv[tid] && ri[tid + off] < ri[tid])) {
          rv[tid] = rv[tid + off]; ri[tid] = ri[tid + off];
        }
      }
      __syncthreads();
    }
    if (tid == 0) {
      idxdown[b * Mm + it] = ri[0];
      sv[ri[0]] = -3.0e38f;
    }
    __syncthreads();
  }
}

// idx_cluster[n] = argmin_m dist[idxdown[m]][n]  (first-min)
__global__ __launch_bounds__(256) void assign_kernel(const float* __restrict__ dist,
                                                     const int* __restrict__ idxdown,
                                                     int* __restrict__ idxc, int b) {
  __shared__ int sidr[Mm];
  int tid = threadIdx.x;
  for (int m = tid; m < Mm; m += 256) sidr[m] = idxdown[b * Mm + m];
  __syncthreads();
  int n = blockIdx.x * 256 + tid;
  if (n < Nn) {
    float best = 3.0e38f; int bmn = 0;
    for (int m = 0; m < Mm; ++m) {
      float d = dist[(size_t)sidr[m] * Nn + n];
      if (d < best) { best = d; bmn = m; }
    }
    idxc[(size_t)b * Nn + n] = bmn;
  }
}

__global__ void override_kernel(const int* __restrict__ idxdown, int* __restrict__ idxc) {
  int g = blockIdx.x * 256 + threadIdx.x;
  if (g < Bn * Mm) {
    int b = g / Mm, m = g % Mm;
    idxc[(size_t)b * Nn + idxdown[g]] = m;
  }
}

__global__ void allw_kernel(const int* __restrict__ idxc, const float* __restrict__ tw,
                            float* __restrict__ allw) {
  int g = blockIdx.x * 256 + threadIdx.x;
  if (g < Bn * Nn) {
    int b = g / Nn;
    atomicAdd(&allw[b * Mm + idxc[g]], tw[g]);
  }
}

__global__ void normw_kernel(const int* __restrict__ idxc, const float* __restrict__ tw,
                             const float* __restrict__ allw, float* __restrict__ normw) {
  int g = blockIdx.x * 256 + threadIdx.x;
  if (g < Bn * Nn) {
    int b = g / Nn;
    normw[g] = tw[g] / (allw[b * Mm + idxc[g]] + 1e-6f);
  }
}

// scatter-mean: down_x[b, c, :] += xn[b, n, :] * normw   (one block per token)
__global__ __launch_bounds__(256) void downx_kernel(const float* __restrict__ xn,
                                                    const int* __restrict__ idxc,
                                                    const float* __restrict__ normw,
                                                    float* __restrict__ downx) {
  int g = blockIdx.x;           // b*Nn + n
  int b = g / Nn;
  int c = idxc[g];
  float wgt = normw[g];
  atomicAdd(&downx[((size_t)(b * Mm + c)) * Dd + threadIdx.x],
            xn[(size_t)g * Dd + threadIdx.x] * wgt);
}

__global__ void aggmul_kernel(const float* __restrict__ aggw, const int* __restrict__ idx_token,
                              const float* __restrict__ normw, float* __restrict__ aggtmp) {
  int g = blockIdx.x * 256 + threadIdx.x;
  if (g < Bn * Nn) {
    int b = g / Nn;
    int t = idx_token[g];
    aggtmp[g] = aggw[g] * normw[(size_t)b * Nn + t];
  }
}

__global__ __launch_bounds__(256) void aggmax_kernel(const float* __restrict__ aggtmp,
                                                     float* __restrict__ aggmax) {
  __shared__ float red[256];
  int b = blockIdx.x, tid = threadIdx.x;
  float m = -3.0e38f;
  for (int n = tid; n < Nn; n += 256) m = fmaxf(m, aggtmp[(size_t)b * Nn + n]);
  float bm = blockReduceMax(m, red, tid);
  if (tid == 0) aggmax[b] = bm;
}

__global__ void aggdiv_kernel(const float* __restrict__ aggtmp, const float* __restrict__ aggmax,
                              float* __restrict__ oagg) {
  int g = blockIdx.x * 256 + threadIdx.x;
  if (g < Bn * Nn) {
    int b = g / Nn;
    oagg[g] = aggtmp[g] / aggmax[b];
  }
}

// 5x5 depthwise stride-2 pad-2 conv; writes f32 + f16 (for the pointwise GEMM)
__global__ __launch_bounds__(256) void dwconv_kernel(const float* __restrict__ xk1,
                                                     const float* __restrict__ dww,
                                                     const float* __restrict__ dwb,
                                                     float* __restrict__ dout,
                                                     _Float16* __restrict__ douth) {
  int r = blockIdx.x;           // b*Pp + p
  int b = r / Pp, p = r % Pp;
  int hp = p / 28, wp = p % 28;
  int d = threadIdx.x;
  float s = dwb[d];
  const float* wrow = dww + d * 25;
#pragma unroll
  for (int kh = 0; kh < 5; ++kh) {
    int h = 2 * hp - 2 + kh;
    if (h < 0 || h >= Hh) continue;
#pragma unroll
    for (int kw = 0; kw < 5; ++kw) {
      int w2 = 2 * wp - 2 + kw;
      if (w2 < 0 || w2 >= Wd) continue;
      s += xk1[((size_t)b * Nn + h * Wd + w2) * Dd + d] * wrow[kh * 5 + kw];
    }
  }
  dout[(size_t)r * Dd + d]  = s;
  douth[(size_t)r * Dd + d] = (_Float16)s;
}

// x_kv = LayerNorm(pwout + x3) with norm1
__global__ __launch_bounds__(256) void ln_kv_kernel(const float* __restrict__ pwout,
                                                    const float* __restrict__ x3,
                                                    const float* __restrict__ g,
                                                    const float* __restrict__ bta,
                                                    float* __restrict__ oxkv) {
  __shared__ float red[256];
  int r = blockIdx.x, tid = threadIdx.x;
  float v  = pwout[(size_t)r * Dd + tid] + x3[(size_t)r * Dd + tid];
  float mu = blockReduceSum(v, red, tid) * (1.f / 256.f);
  float dv = v - mu;
  float var = blockReduceSum(dv * dv, red, tid) * (1.f / 256.f);
  oxkv[(size_t)r * Dd + tid] = dv * rsqrtf(var + 1e-5f) * g[tid] + bta[tid];
}

// ---------------------------------------------------------------------------
// Host orchestration
// ---------------------------------------------------------------------------
extern "C" void kernel_launch(void* const* d_in, const int* in_sizes, int n_in,
                              void* d_out, int out_size, void* d_ws, size_t ws_size,
                              hipStream_t stream) {
  (void)in_sizes; (void)n_in; (void)out_size; (void)ws_size;

  const float* x           = (const float*)d_in[0];
  const int*   idx_token   = (const int*)d_in[1];
  const float* agg_weight  = (const float*)d_in[2];
  const float* conv_skip_w = (const float*)d_in[3];
  const float* conv_w      = (const float*)d_in[4];
  const float* conv_b      = (const float*)d_in[5];
  const float* ms_skip_w   = (const float*)d_in[6];
  const float* ms_dw_w     = (const float*)d_in[7];
  const float* ms_dw_b     = (const float*)d_in[8];
  const float* ms_pw_w     = (const float*)d_in[9];
  const float* ms_pw_b     = (const float*)d_in[10];
  const float* norm_g      = (const float*)d_in[11];
  const float* norm_b      = (const float*)d_in[12];
  const float* norm1_g     = (const float*)d_in[13];
  const float* norm1_b     = (const float*)d_in[14];
  const float* score_w     = (const float*)d_in[15];
  const float* score_b     = (const float*)d_in[16];

  float* out     = (float*)d_out;
  float* o_downx = out;                                  // B*M*D
  float* o_xkv   = o_downx + (size_t)Bn * Mm * Dd;       // B*P*D
  float* o_score = o_xkv   + (size_t)Bn * Pp * Dd;       // B*N
  float* o_agg   = o_score + (size_t)Bn * Nn;            // B*N

  // ---- workspace carve-out ----
  char* wsb = (char*)d_ws;
  size_t cur = 0;
  auto carve = [&](size_t bytes) -> char* {
    char* p = wsb + cur;
    cur = (cur + bytes + 255) & ~(size_t)255;
    return p;
  };
  float*     ws_xn      = (float*)    carve((size_t)Bn * Nn * Dd * 4);
  _Float16*  ws_xnh     = (_Float16*) carve((size_t)Bn * Nn * Dd * 2);
  float*     ws_x3      = (float*)    carve((size_t)Bn * Pp * Dd * 4);
  float*     ws_xsk     = (float*)    carve((size_t)Bn * Nn * Dd * 4);  // reused as xk1
  float*     ws_pwout   = (float*)    carve((size_t)Bn * Pp * Dd * 4);
  _Float16*  ws_col     = (_Float16*) carve((size_t)Bn * Pp * K9 * 2);
  _Float16*  ws_xh      = (_Float16*) carve((size_t)Bn * Nn * Ee * 2);
  _Float16*  ws_wskip   = (_Float16*) carve((size_t)Dd * Ee * 2);
  _Float16*  ws_wconv   = (_Float16*) carve((size_t)Dd * K9 * 2);
  _Float16*  ws_wms     = (_Float16*) carve((size_t)Dd * Ee * 2);
  _Float16*  ws_wpw     = (_Float16*) carve((size_t)Dd * Dd * 2);
  float*     ws_dwout   = (float*)    carve((size_t)Bn * Pp * Dd * 4);
  _Float16*  ws_dwouth  = (_Float16*) carve((size_t)Bn * Pp * Dd * 2);
  float*     ws_dist    = (float*)    carve((size_t)Nn * Nn * 4);
  float*     ws_x2      = (float*)    carve((size_t)Bn * Nn * 4);
  float*     ws_dens    = (float*)    carve((size_t)Bn * Nn * 4);
  float*     ws_score   = (float*)    carve((size_t)Bn * Nn * 4);
  float*     ws_tw      = (float*)    carve((size_t)Bn * Nn * 4);
  float*     ws_normw   = (float*)    carve((size_t)Bn * Nn * 4);
  float*     ws_aggtmp  = (float*)    carve((size_t)Bn * Nn * 4);
  float*     ws_allw    = (float*)    carve((size_t)Bn * Mm * 4);
  float*     ws_dmax    = (float*)    carve((size_t)Bn * 4);
  float*     ws_aggmax  = (float*)    carve((size_t)Bn * 4);
  int*       ws_idxdown = (int*)      carve((size_t)Bn * Mm * 4);
  int*       ws_idxc    = (int*)      carve((size_t)Bn * Nn * 4);

  const dim3 blk(256);
  auto cdiv = [](int a, int b) { return (a + b - 1) / b; };
  auto gemm = [&](const _Float16* A, const _Float16* Bm, const float* bias, float* C,
                  int Mt, int Nt, int Kt) {
    dim3 g(cdiv(Nt, 64), cdiv(Mt, 128));
    wmma_gemm_kernel<<<g, blk, 0, stream>>>(A, Bm, bias, C, Mt, Nt, Kt);
  };

  // ---- init (must re-zero every call: atomics accumulate) ----
  zerof_kernel<<<cdiv(Bn * Mm * Dd, 256), blk, 0, stream>>>(o_downx, Bn * Mm * Dd);
  zerof_kernel<<<cdiv(Bn * Mm, 256),      blk, 0, stream>>>(ws_allw, Bn * Mm);
  zerof_kernel<<<1,                        blk, 0, stream>>>(ws_dmax, Bn);

  // ---- f16 operand staging ----
  cvt_f32_f16_kernel<<<cdiv(Bn * Nn * Ee, 256), blk, 0, stream>>>(x, ws_xh, Bn * Nn * Ee);
  cvt_f32_f16_kernel<<<cdiv(Dd * Ee, 256),      blk, 0, stream>>>(conv_skip_w, ws_wskip, Dd * Ee);
  cvt_f32_f16_kernel<<<cdiv(Dd * K9, 256),      blk, 0, stream>>>(conv_w, ws_wconv, Dd * K9);
  cvt_f32_f16_kernel<<<cdiv(Dd * Ee, 256),      blk, 0, stream>>>(ms_skip_w, ws_wms, Dd * Ee);
  cvt_f32_f16_kernel<<<cdiv(Dd * Dd, 256),      blk, 0, stream>>>(ms_pw_w, ws_wpw, Dd * Dd);

  // ---- token skip GEMM + conv-as-GEMM ----
  gemm(ws_xh, ws_wskip, nullptr, ws_xsk, Bn * Nn, Dd, Ee);
  im2col_kernel<<<Bn * Pp, blk, 0, stream>>>(x, ws_col);
  gemm(ws_col, ws_wconv, conv_b, ws_x3, Bn * Pp, Dd, K9);

  // ---- xc -> LayerNorm -> xn / token_score / token_weight ----
  ln_xc_kernel<<<Bn * Nn, blk, 0, stream>>>(ws_xsk, ws_x3, norm_g, norm_b,
                                            score_w, score_b, ws_xn, ws_xnh, o_score, ws_tw);
  x2_kernel<<<Bn * Nn, blk, 0, stream>>>(ws_xn, ws_x2);

  // ---- DPC-KNN per batch (dist matrix materialized per batch in ws) ----
  for (int b = 0; b < Bn; ++b) {
    const _Float16* xhb = ws_xnh + (size_t)b * Nn * Dd;
    const float*    x2b = ws_x2 + (size_t)b * Nn;
    dim3 gd(cdiv(Nn, 64), cdiv(Nn, 128));
    wmma_dist_kernel<<<gd, blk, 0, stream>>>(xhb, x2b, ws_dist, ws_dmax + b);
    density_kernel<<<Nn, blk, 0, stream>>>(ws_dist, ws_dens, b);
    dmin_kernel<<<Nn, blk, 0, stream>>>(ws_dist, ws_dens, ws_dmax + b, ws_score, b);
    topm_kernel<<<1, blk, 0, stream>>>(ws_score, ws_idxdown, b);
    assign_kernel<<<cdiv(Nn, 256), blk, 0, stream>>>(ws_dist, ws_idxdown, ws_idxc, b);
  }
  override_kernel<<<cdiv(Bn * Mm, 256), blk, 0, stream>>>(ws_idxdown, ws_idxc);

  // ---- merge tokens ----
  allw_kernel<<<cdiv(Bn * Nn, 256), blk, 0, stream>>>(ws_idxc, ws_tw, ws_allw);
  normw_kernel<<<cdiv(Bn * Nn, 256), blk, 0, stream>>>(ws_idxc, ws_tw, ws_allw, ws_normw);
  downx_kernel<<<Bn * Nn, blk, 0, stream>>>(ws_xn, ws_idxc, ws_normw, o_downx);
  aggmul_kernel<<<cdiv(Bn * Nn, 256), blk, 0, stream>>>(agg_weight, idx_token, ws_normw, ws_aggtmp);
  aggmax_kernel<<<Bn, blk, 0, stream>>>(ws_aggtmp, ws_aggmax);
  aggdiv_kernel<<<cdiv(Bn * Nn, 256), blk, 0, stream>>>(ws_aggtmp, ws_aggmax, o_agg);

  // ---- multiscale branch ----
  gemm(ws_xh, ws_wms, nullptr, ws_xsk, Bn * Nn, Dd, Ee);        // xk1 reuses xsk buffer
  dwconv_kernel<<<Bn * Pp, blk, 0, stream>>>(ws_xsk, ms_dw_w, ms_dw_b, ws_dwout, ws_dwouth);
  gemm(ws_dwouth, ws_wpw, ms_pw_b, ws_pwout, Bn * Pp, Dd, Dd);
  ln_kv_kernel<<<Bn * Pp, blk, 0, stream>>>(ws_pwout, ws_x3, norm1_g, norm1_b, o_xkv);
}
